// Transformer_84396107366685
// MI455X (gfx1250) — matrix operand order repaired and checked by reading, not compiled, not run on hardware
//
#include <hip/hip_runtime.h>
#include <hip/hip_bf16.h>

typedef __attribute__((ext_vector_type(16))) _Float16 v16h;
typedef __attribute__((ext_vector_type(8)))  float    v8f;

#define B_    16
#define T_    1024
#define TKV_  1023
#define D_IN_ 384
#define M_    832
#define EMB_  256
#define KS_   64
#define HID_  1024
#define OUT_  32
#define IN_   64
#define NL_   3
#define EPS_  1e-6f

__device__ inline float gelu_tanh(float x) {
  const float c0 = 0.7978845608028654f; // sqrt(2/pi)
  return 0.5f * x * (1.0f + tanhf(c0 * (x + 0.044715f * x * x * x)));
}

// v_cvt_pk_rtz_f16_f32: two f32 -> packed 2xf16 in one dword (a=low, b=high)
__device__ inline unsigned pack2h(float a, float b) {
  union { decltype(__builtin_amdgcn_cvt_pkrtz(0.f, 0.f)) h; unsigned u; } c;
  c.h = __builtin_amdgcn_cvt_pkrtz(a, b);
  return c.u;
}

// ---------------------------------------------------------------------------
// Block = 4 waves (32,4). C macro-tile 128(M) x 64(N), K-step 32.
// Wave tile 32x64: 2 A frags x 4 B frags -> 8 v_wmma_f32_16x16x32_f16 / K-step.
// LDS layouts:
//   A : [row][k'] halves, pitch PA=40  -> frag = 2x ds_load_b128
//   B NN: [kpair][n] packed dwords, pitch PBW=68 -> frag = 8x ds_load_b32
//   B NT: [n][k'] halves, pitch PA    -> frag = 2x ds_load_b128
// Staging paths (all selected by UNIFORM predicates):
//   1) aligned + full tile : float4 loads, no masks
//   2) unaligned + full    : plain b32 loads, no masks (vec=att@v interior)
//   3) edge/tail           : clamped addresses + integer-AND value masking
template<bool BT, bool RESID, bool GELU>
__global__ __launch_bounds__(128) void gemm_wmma(
    const float* __restrict__ A,  long long lda, long long sA,
    const float* __restrict__ Bm, long long ldb, long long sB,
    const float* __restrict__ bias,
    float* __restrict__ C,        long long ldc, long long sC,
    int rows, int ncols, int K, int A4)
{
  constexpr int PA  = 40;                     // pitch (halves)
  constexpr int PBW = 68;                     // pitch (dwords) for NN packed B
  constexpr int BDW = BT ? (64 * PA / 2) : (16 * PBW);   // B LDS size in dwords
  __shared__ alignas(16) _Float16 As[128 * PA];
  __shared__ alignas(16) unsigned Bsw[BDW];

  long long z = blockIdx.z;
  A  += z * sA;
  Bm += z * sB;
  C  += z * sC;
  int lane = threadIdx.x;
  int wv   = threadIdx.y;               // 0..3 -> 32-row M sub-tile
  int tid  = wv * 32 + lane;            // 0..127
  int tileM = blockIdx.x * 128;
  int tileN = blockIdx.y * 64;

  // A staging: one thread per A row (32 k-values)
  int amr  = tid;
  int gmA  = tileM + amr;
  bool aok = (gmA < rows);
  unsigned amask = aok ? 0xFFFFFFFFu : 0u;
  long long rowA = (long long)(aok ? gmA : (rows - 1)) * lda;
  bool mFull = (tileM + 128 <= rows);         // uniform
  // B staging (NN): thread -> (kpair bq, 8-wide n segment)
  int bq  = tid >> 3;                   // 0..15
  int bns = (tid & 7) << 3;             // 0..56
  // B staging (NT): thread -> (n row, 16-wide k segment)
  int bnr = tid >> 1;
  int bks = (tid & 1) << 4;
  bool nFull = (tileN + 64 <= ncols);         // uniform

  v8f acc0[4] = {};
  v8f acc1[4] = {};

  for (int k0 = 0; k0 < K; k0 += 32) {
    bool kFull = (k0 + 32 <= K);              // uniform
    // ---------------- stage A tile (128 x 32) ----------------
    {
      unsigned hw[16];
      if (kFull && mFull && A4) {             // uniform: aligned fast path
        const float4* p4 = (const float4*)(A + rowA + k0);
        __builtin_prefetch(A + rowA + k0 + 32, 0, 1);
#pragma unroll
        for (int g = 0; g < 8; ++g) {
          float4 f = p4[g];
          hw[2*g+0] = pack2h(f.x, f.y);
          hw[2*g+1] = pack2h(f.z, f.w);
        }
      } else if (kFull && mFull) {            // uniform: unaligned but full tile
        const float* Af = A + rowA + k0;      // (vec = att@v interior, lda%4!=0)
#pragma unroll
        for (int i = 0; i < 16; ++i) hw[i] = pack2h(Af[2*i], Af[2*i + 1]);
      } else {                                // edge/tail: clamped + masked
        const unsigned* Au = (const unsigned*)A;
#pragma unroll
        for (int i = 0; i < 16; ++i) {
          int kg0 = k0 + 2*i, kg1 = kg0 + 1;
          int kc0 = (kg0 < K) ? kg0 : (K - 1);
          int kc1 = (kg1 < K) ? kg1 : (K - 1);
          unsigned u0 = Au[rowA + kc0] & ((kg0 < K) ? amask : 0u);
          unsigned u1 = Au[rowA + kc1] & ((kg1 < K) ? amask : 0u);
          hw[i] = pack2h(__uint_as_float(u0), __uint_as_float(u1));
        }
      }
      int4* d = (int4*)&As[amr * PA];
#pragma unroll
      for (int g = 0; g < 4; ++g) d[g] = ((const int4*)hw)[g];
    }
    // ---------------- stage B tile (32 x 64) ----------------
    if (!BT) {
      // B stored K x N -> LDS [kpair][n] packed dwords
      unsigned hw[8];
      int gk0 = k0 + 2*bq, gk1 = gk0 + 1;
      int gn  = tileN + bns;
      long long rowB0 = (long long)((gk0 < K) ? gk0 : (K - 1)) * ldb;
      long long rowB1 = (long long)((gk1 < K) ? gk1 : (K - 1)) * ldb;
      if (kFull && nFull) {                   // uniform, unguarded
        const float4* p0 = (const float4*)(Bm + rowB0 + gn);
        const float4* p1 = (const float4*)(Bm + rowB1 + gn);
        float4 f0 = p0[0], f1 = p0[1];
        float4 g0 = p1[0], g1 = p1[1];
        hw[0] = pack2h(f0.x, g0.x); hw[1] = pack2h(f0.y, g0.y);
        hw[2] = pack2h(f0.z, g0.z); hw[3] = pack2h(f0.w, g0.w);
        hw[4] = pack2h(f1.x, g1.x); hw[5] = pack2h(f1.y, g1.y);
        hw[6] = pack2h(f1.z, g1.z); hw[7] = pack2h(f1.w, g1.w);
      } else {
        const unsigned* Bu = (const unsigned*)Bm;
        unsigned km0 = (gk0 < K) ? 0xFFFFFFFFu : 0u;
        unsigned km1 = (gk1 < K) ? 0xFFFFFFFFu : 0u;
#pragma unroll
        for (int i = 0; i < 8; ++i) {
          int ng = gn + i;
          int nc = (ng < ncols) ? ng : (ncols - 1);
          unsigned nm = (ng < ncols) ? 0xFFFFFFFFu : 0u;
          unsigned u0 = Bu[rowB0 + nc] & (km0 & nm);
          unsigned u1 = Bu[rowB1 + nc] & (km1 & nm);
          hw[i] = pack2h(__uint_as_float(u0), __uint_as_float(u1));
        }
      }
      int4* d = (int4*)&Bsw[bq * PBW + bns];
      d[0] = ((const int4*)hw)[0];
      d[1] = ((const int4*)hw)[1];
    } else {
      // B stored N x K -> LDS [n][k'] halves, pitch PA
      unsigned hw[8];
      int gn  = tileN + bnr;
      bool nok = (gn < ncols);
      long long rowB = (long long)(nok ? gn : (ncols - 1)) * ldb;
      if (kFull && nFull) {                   // uniform, unguarded
        const float4* p4 = (const float4*)(Bm + rowB + k0 + bks);
#pragma unroll
        for (int g = 0; g < 4; ++g) {
          float4 f = p4[g];
          hw[2*g+0] = pack2h(f.x, f.y);
          hw[2*g+1] = pack2h(f.z, f.w);
        }
      } else {
        const unsigned* Bu = (const unsigned*)Bm;
        unsigned nmask = nok ? 0xFFFFFFFFu : 0u;
#pragma unroll
        for (int i = 0; i < 8; ++i) {
          int kg0 = k0 + bks + 2*i, kg1 = kg0 + 1;
          int kc0 = (kg0 < K) ? kg0 : (K - 1);
          int kc1 = (kg1 < K) ? kg1 : (K - 1);
          unsigned u0 = Bu[rowB + kc0] & ((kg0 < K) ? nmask : 0u);
          unsigned u1 = Bu[rowB + kc1] & ((kg1 < K) ? nmask : 0u);
          hw[i] = pack2h(__uint_as_float(u0), __uint_as_float(u1));
        }
      }
      _Float16* Bsh = (_Float16*)Bsw;
      int4* d = (int4*)&Bsh[bnr * PA + bks];
      d[0] = ((const int4*)hw)[0];
      d[1] = ((const int4*)hw)[1];
    }
    __syncthreads();

    // ---------------- compute: 8 WMMAs per wave ----------------
    {
      // A fragment (16-bit A 16x32): lane<16 -> K {0..15},
      // lane>=16 -> K {16..31}: two contiguous 16B runs per lane.
      int kb = (lane & 16) ? 16 : 0;
      const _Float16* ap0 = &As[(wv * 32 + (lane & 15)) * PA];
      const _Float16* ap1 = ap0 + 16 * PA;
      union { int4 q[2]; v16h v; } ua0, ua1;
      ua0.q[0] = *(const int4*)(ap0 + kb);
      ua0.q[1] = *(const int4*)(ap0 + 8 + kb);
      ua1.q[0] = *(const int4*)(ap1 + kb);
      ua1.q[1] = *(const int4*)(ap1 + 8 + kb);

      int nl  = lane & 15;
#pragma unroll
      for (int nt = 0; nt < 4; ++nt) {
        v16h b;
        if (BT) {
          int kb2 = (lane & 16) ? 16 : 0;
          const _Float16* Bsh = (const _Float16*)Bsw;
          union { int4 q[2]; v16h v; } ub;
          const _Float16* bp = &Bsh[(nt * 16 + nl) * PA + kb2];
          ub.q[0] = ((const int4*)bp)[0];
          ub.q[1] = ((const int4*)bp)[1];
          b = ub.v;
        } else {
          int kp = (lane & 16) ? 8 : 0;       // kpair base
          union { unsigned w[8]; v16h v; } ub;
#pragma unroll
          for (int t = 0; t < 8; ++t) ub.w[t] = Bsw[(kp + t) * PBW + nt * 16 + nl];
          b = ub.v;
        }
        acc0[nt] = __builtin_amdgcn_wmma_f32_16x16x32_f16(
            false, ua0.v, false, b, (short)0, acc0[nt], false, false);
        acc1[nt] = __builtin_amdgcn_wmma_f32_16x16x32_f16(
            false, ua1.v, false, b, (short)0, acc1[nt], false, false);
      }
    }
    __syncthreads();
  }

  // ---------------- epilogue ----------------
  int nl     = lane & 15;
  int rbase0 = tileM + wv * 32 + ((lane & 16) ? 8 : 0);
  int rbase1 = rbase0 + 16;
#pragma unroll
  for (int nt = 0; nt < 4; ++nt) {
    int col  = tileN + nt * 16 + nl;
    int colc = (col < ncols) ? col : (ncols - 1);
    bool cok = (col < ncols);
    float bv = bias ? bias[colc] : 0.0f;
#pragma unroll
    for (int r = 0; r < 8; ++r) {
      int row0 = rbase0 + r;
      if (cok && row0 < rows) {
        float v = acc0[nt][r] + bv;
        if (GELU) v = gelu_tanh(v);
        float* Cp = C + (long long)row0 * ldc + col;
        if (RESID) v += *Cp;
        *Cp = v;
      }
      int row1 = rbase1 + r;
      if (cok && row1 < rows) {
        float v = acc1[nt][r] + bv;
        if (GELU) v = gelu_tanh(v);
        float* Cp = C + (long long)row1 * ldc + col;
        if (RESID) v += *Cp;
        *Cp = v;
      }
    }
  }
}

// ---------------------------------------------------------------------------
// q / (sum(q, head) + eps): one thread per (row, head), head size 64.
__global__ void norm_heads(float* __restrict__ p, int rows, int heads, long long ld) {
  long long idx = (long long)blockIdx.x * blockDim.x + threadIdx.x;
  if (idx >= (long long)rows * heads) return;
  long long r = idx / heads;
  int hh = (int)(idx % heads);
  float* q = p + r * ld + (long long)hh * KS_;
  float s = 0.0f;
#pragma unroll 8
  for (int i = 0; i < KS_; ++i) s += q[i];
  float d = s + EPS_;
#pragma unroll 8
  for (int i = 0; i < KS_; ++i) q[i] /= d;
}

// W_e = emb_w[IN:IN+OUT, IN:IN+EMB]
__global__ void extract_We(const float* __restrict__ emb_w, float* __restrict__ We) {
  int i = blockIdx.x;   // 0..31
  int j = threadIdx.x;  // 0..255
  We[i * EMB_ + j] = emb_w[(IN_ + i) * M_ + (IN_ + j)];
}

// Row softmax over 32 logits; one wave32 per row; shuffle reductions.
__global__ void softmax_rows(const float* __restrict__ L,
                             float* __restrict__ o1, float* __restrict__ o2) {
  int r = blockIdx.x;
  int lane = threadIdx.x;   // 0..31
  float v = L[r * OUT_ + lane];
  float m = v;
  for (int o = 16; o > 0; o >>= 1) m = fmaxf(m, __shfl_xor(m, o, 32));
  float e = __expf(v - m);
  float s = e;
  for (int o = 16; o > 0; o >>= 1) s += __shfl_xor(s, o, 32);
  float out = e / s;
  o1[r * OUT_ + lane] = out;
  if (o2) o2[r * OUT_ + lane] = out;
}

// ---------------------------------------------------------------------------
extern "C" void kernel_launch(void* const* d_in, const int* in_sizes, int n_in,
                              void* d_out, int out_size, void* d_ws, size_t ws_size,
                              hipStream_t stream) {
  (void)in_sizes; (void)n_in; (void)out_size; (void)ws_size;
  const float* x     = (const float*)d_in[0];
  const float* emb_w = (const float*)d_in[1];
  const float* qf_w  = (const float*)d_in[2];
  const float* qf_b  = (const float*)d_in[3];
  const float* kf_w  = (const float*)d_in[4];
  const float* kf_b  = (const float*)d_in[5];
  const float* vf_w  = (const float*)d_in[6];
  const float* vf_b  = (const float*)d_in[7];
  const float* pf_w  = (const float*)d_in[8];
  const float* pf_b  = (const float*)d_in[9];
  const float* qe_w  = (const float*)d_in[10];
  const float* qe_b  = (const float*)d_in[11];
  const float* ke_w  = (const float*)d_in[12];
  const float* ke_b  = (const float*)d_in[13];
  const float* ve_w  = (const float*)d_in[14];
  const float* ve_b  = (const float*)d_in[15];
  const float* pe_w  = (const float*)d_in[16];
  const float* pe_b  = (const float*)d_in[17];
  const float* m1_w  = (const float*)d_in[18];
  const float* m1_b  = (const float*)d_in[19];
  const float* m2_w  = (const float*)d_in[20];
  const float* m2_b  = (const float*)d_in[21];

  // Output layout: out(16,32) | stack_h(3,16,32) | stack_f(3,16,2,1024,1023)
  float* out = (float*)d_out;
  float* sh  = out + 512;
  float* sf  = out + 2048;
  const long long TT = (long long)T_ * TKV_;

  // Workspace (~122 MB)
  const long long R = (long long)B_ * T_;   // 16384 rows
  float* ws    = (float*)d_ws;
  float* h     = ws;                        // R*832
  float* base2 = ws + R * M_;
  const long long QKV = R * 128;
  float* qf   = base2;
  float* kf   = base2 + QKV;
  float* vf   = base2 + 2 * QKV;
  float* vec  = base2 + 3 * QKV;
  float* qe   = base2;                      // reuse after fwd-attn
  float* atte = base2 + R * KS_;
  float* vece = atte  + R * OUT_;
  float* hid  = base2;                      // reuse during MLP
  float* pers = base2 + R * HID_;
  float* We     = pers;                     // 32*256
  float* keb    = We  + OUT_ * EMB_;        // 32*64 (normalized)
  float* veb    = keb + OUT_ * KS_;         // 32*64
  float* logits = veb + OUT_ * KS_;         // 16*32

#define GEMM(BT, RESID, GELU, A, lda, sA, Bm, ldb, sB, bias, C, ldc, sC, rows, ncols, K, Z, A4) \
  do { dim3 g(((rows) + 127) / 128, ((ncols) + 63) / 64, (Z)); dim3 blk(32, 4, 1);              \
       gemm_wmma<BT, RESID, GELU><<<g, blk, 0, stream>>>(                                       \
           A, lda, sA, Bm, ldb, sB, bias, C, ldc, sC, rows, ncols, K, A4); } while (0)

  // h = x @ emb_w
  GEMM(false, false, false, x, D_IN_, 0, emb_w, M_, 0, nullptr, h, M_, 0, (int)R, M_, D_IN_, 1, 1);

  // W_e and fixed embedding-attention K/V (layer-invariant)
  extract_We<<<dim3(OUT_), dim3(EMB_), 0, stream>>>(emb_w, We);
  GEMM(false, false, false, We, EMB_, 0, ke_w, KS_, 0, ke_b, keb, KS_, 0, OUT_, KS_, EMB_, 1, 1);
  norm_heads<<<1, 32, 0, stream>>>(keb, OUT_, 1, KS_);
  GEMM(false, false, false, We, EMB_, 0, ve_w, KS_, 0, ve_b, veb, KS_, 0, OUT_, KS_, EMB_, 1, 1);

  for (int l = 0; l < NL_; ++l) {
    float* attL = sf + (long long)l * B_ * 2 * TT;

    // ---- forward linear attention (H=2, KV = first 1023 tokens per batch)
    GEMM(false, false, false, h, M_, 0, qf_w, 128, 0, qf_b, qf, 128, 0, (int)R, 128, M_, 1, 1);
    GEMM(false, false, false, h, M_, 0, kf_w, 128, 0, kf_b, kf, 128, 0, (int)R, 128, M_, 1, 1);
    GEMM(false, false, false, h, M_, 0, vf_w, 128, 0, vf_b, vf, 128, 0, (int)R, 128, M_, 1, 1);
    {
      int nt = (int)R * 2;
      norm_heads<<<(nt + 255) / 256, 256, 0, stream>>>(qf, (int)R, 2, 128);
      norm_heads<<<(nt + 255) / 256, 256, 0, stream>>>(kf, (int)R, 2, 128);
    }
    for (int hh = 0; hh < 2; ++hh) {
      // att[b,hh] = q(1024x64) @ k(1023x64)^T  -> straight into stack_f
      GEMM(true, false, false,
           qf + hh * KS_, 128, (long long)T_ * 128,
           kf + hh * KS_, 128, (long long)T_ * 128, nullptr,
           attL + hh * TT, TKV_, 2 * TT, T_, TKV_, KS_, B_, 1);
      // vec[b,hh] = att(1024x1023) @ v(1023x64)  (lda=1023 -> unaligned-b32 path)
      GEMM(false, false, false,
           attL + hh * TT, TKV_, 2 * TT,
           vf + hh * KS_, 128, (long long)T_ * 128, nullptr,
           vec + hh * KS_, 128, (long long)T_ * 128, T_, KS_, TKV_, B_, 0);
    }
    // h += vec @ pf_w + pf_b
    GEMM(false, true, false, vec, 128, 0, pf_w, M_, 0, pf_b, h, M_, 0, (int)R, M_, 128, 1, 1);

    // ---- embedding attention (H=1, K/V from fixed W_e)
    GEMM(false, false, false, h, M_, 0, qe_w, KS_, 0, qe_b, qe, KS_, 0, (int)R, KS_, M_, 1, 1);
    norm_heads<<<((int)R + 255) / 256, 256, 0, stream>>>(qe, (int)R, 1, KS_);
    GEMM(true,  false, false, qe, KS_, 0, keb, KS_, 0, nullptr, atte, OUT_, 0, (int)R, OUT_, KS_, 1, 1);
    GEMM(false, false, false, atte, OUT_, 0, veb, KS_, 0, nullptr, vece, KS_, 0, (int)R, KS_, OUT_, 1, 1);
    GEMM(false, true,  false, vece, KS_, 0, pe_w, M_, 0, pe_b, h, M_, 0, (int)R, M_, KS_, 1, 1);

    // ---- MLP on last EMB slice of h (in-place residual)
    GEMM(false, false, true,  h + (M_ - EMB_), M_, 0, m1_w, HID_, 0, m1_b, hid, HID_, 0,
         (int)R, HID_, EMB_, 1, 1);
    GEMM(false, true,  false, hid, HID_, 0, m2_w, EMB_, 0, m2_b, h + (M_ - EMB_), M_, 0,
         (int)R, EMB_, HID_, 1, 1);

    // ---- logits = f @ W_e^T ; f = h[:, T-1, -EMB:] (row stride T*M per batch)
    GEMM(true, false, false,
         h + (long long)(T_ - 1) * M_ + (M_ - EMB_), (long long)T_ * M_, 0,
         We, EMB_, 0, nullptr, logits, OUT_, 0, B_, OUT_, EMB_, 1, 1);
    softmax_rows<<<B_, 32, 0, stream>>>(logits, sh + (long long)l * B_ * OUT_,
                                        (l == NL_ - 1) ? out : nullptr);
  }
#undef GEMM
}
